// MultiHeadAttention_575525617984
// MI455X (gfx1250) — compile-verified, matrix operand order checked by reading, not run on hardware
//
#include <hip/hip_runtime.h>

typedef __bf16 bf16_t;
typedef __attribute__((ext_vector_type(8)))  __bf16 bf16x8;
typedef __attribute__((ext_vector_type(16))) __bf16 bf16x16;
typedef __attribute__((ext_vector_type(8)))  float  f32x8;
typedef __attribute__((ext_vector_type(4)))  unsigned int u32x4;
typedef __attribute__((ext_vector_type(8)))  unsigned int u32x8;

#define B_  2
#define S_  4096
#define D_  512
#define H_  8
#define DK_ 64
#define M_  (B_ * S_)   // 8192
#define KDIM_ 512

// LDS strides (elements). 40*2=80B, 72*2=144B: both multiples of 16B for b128 ops,
// and both reachable with TDM pad_interval/pad_amount (64B+16B, 128B+16B).
#define LDSA 40
#define LDSK 72

__device__ __forceinline__ bf16x16 make_frag(const bf16_t* lo, const bf16_t* hi) {
    bf16x8 a = *(const bf16x8*)lo;
    bf16x8 b = *(const bf16x8*)hi;
    return __builtin_shufflevector(a, b, 0,1,2,3,4,5,6,7,8,9,10,11,12,13,14,15);
}

__device__ __forceinline__ f32x8 wmma_bf16(bf16x16 a, bf16x16 b, f32x8 c) {
    return __builtin_amdgcn_wmma_f32_16x16x32_bf16(false, a, false, b, (short)0, c,
                                                   false, false);
}

// ---------------------------------------------------------------------------
// Tensor Data Mover: 2D tile (tile1 rows x tile0 elems of 2-byte data) from a
// row-major tensor (dim1 x dim0, row stride stride0 elems) into LDS at
// lds_byte_off, inserting pad_amount(+1) DWORDs of pad every 2<<pad_interval
// DWORDs (CDNA5 ISA sect. 8).  Issued per-wave; tracked by
// TENSORcnt; EXEC ignored.
// ---------------------------------------------------------------------------
__device__ __forceinline__ void tdm_load_2d(unsigned lds_byte_off, const void* gaddr,
                                            unsigned tile0, unsigned tile1,
                                            unsigned dim0, unsigned dim1,
                                            unsigned stride0,
                                            unsigned pad_interval, unsigned pad_amount) {
    const unsigned long long ga = (unsigned long long)gaddr;
    u32x4 g0;
    g0[0] = 1u;                                              // count=1, user mode
    g0[1] = lds_byte_off;                                    // lds_addr [63:32]
    g0[2] = (unsigned)ga;                                    // global_addr lo
    g0[3] = (unsigned)((ga >> 32) & 0x1FFFFFFu) | (2u << 30); // addr hi | type=2
    u32x8 g1;
    g1[0] = (1u << 16)                      // data_size = 1 -> 2 bytes
          | (1u << 20)                      // pad_enable
          | (pad_interval << 22)
          | (pad_amount << 25);
    g1[1] = (dim0 & 0xFFFFu) << 16;                          // tensor_dim0 lo16
    g1[2] = ((dim0 >> 16) & 0xFFFFu) | ((dim1 & 0xFFFFu) << 16);
    g1[3] = ((dim1 >> 16) & 0xFFFFu) | (tile0 << 16);        // tile_dim0
    g1[4] = tile1;                                           // tile_dim1 (tile_dim2=0)
    g1[5] = stride0;                                         // tensor_dim0_stride lo32
    g1[6] = 0u;
    g1[7] = 0u;
    asm volatile("tensor_load_to_lds %0, %1" :: "s"(g0), "s"(g1) : "memory");
}

// ---------------------------------------------------------------------------
// GEMM body: C[M,N] = A[M,K] @ W[N,K]^T + bias,  M=8192, N=512, K=512
// Block: 256 thr (8 waves), tile 128x64, BK=32.
// A_BF16 path stages the A tile with the TDM (wave 0), f32 path converts in VALU.
// ---------------------------------------------------------------------------
template <bool A_BF16, bool SPLIT_HEADS>
__device__ void gemm_body(const void* __restrict__ Ap,
                          const float* __restrict__ W,
                          const float* __restrict__ bias,
                          void* __restrict__ Out,
                          bf16_t* As, bf16_t* Bs) {
    const int tid   = threadIdx.x;
    const int wid   = tid >> 5;
    const int lane  = tid & 31;
    const int laneN = lane & 15;
    const int laneH = lane >> 4;      // 0/1
    const int kb    = laneH * 8;      // per-half K base within fragment
    const int bm    = blockIdx.x * 128;
    const int bn    = blockIdx.y * 64;

    f32x8 acc[4] = {};

    for (int k0 = 0; k0 < KDIM_; k0 += 32) {
        // --- stage A tile 128x32 ---
        if (A_BF16) {
            if (wid == 0) {
                const bf16_t* src = (const bf16_t*)Ap + (size_t)bm * KDIM_ + k0;
                // rows of 32 elems (64B = 16 DW): pad_interval code 3, pad 4 DW (16B)
                tdm_load_2d((unsigned)(size_t)(void*)As, src,
                            /*tile0=*/32, /*tile1=*/128,
                            /*dim0=*/KDIM_, /*dim1=*/M_, /*stride0=*/KDIM_,
                            /*pad_interval=*/3, /*pad_amount=*/3);
                __builtin_amdgcn_s_wait_tensorcnt(0);
            }
        } else {
            const int row = tid >> 1, c = (tid & 1) * 16;
            const float* src = (const float*)Ap + (size_t)(bm + row) * KDIM_ + k0 + c;
            bf16x8 lo, hi;
#pragma unroll
            for (int i = 0; i < 8; ++i) { lo[i] = (bf16_t)src[i]; hi[i] = (bf16_t)src[8 + i]; }
            *(bf16x8*)&As[row * LDSA + c]     = lo;
            *(bf16x8*)&As[row * LDSA + c + 8] = hi;
        }
        // --- stage W tile 64x32 ---
        {
            const int row = tid >> 2, c = (tid & 3) * 8;
            const float* src = W + (size_t)(bn + row) * KDIM_ + k0 + c;
            bf16x8 v;
#pragma unroll
            for (int i = 0; i < 8; ++i) v[i] = (bf16_t)src[i];
            *(bf16x8*)&Bs[row * LDSA + c] = v;
        }
        __syncthreads();

        const int arow = wid * 16 + laneN;
        bf16x16 a = make_frag(&As[arow * LDSA + kb], &As[arow * LDSA + kb + 16]);
#pragma unroll
        for (int j = 0; j < 4; ++j) {
            const int brow = j * 16 + laneN;
            bf16x16 b = make_frag(&Bs[brow * LDSA + kb], &Bs[brow * LDSA + kb + 16]);
            acc[j] = wmma_bf16(a, b, acc[j]);
        }
        __syncthreads();
    }

    // --- epilogue ---
#pragma unroll
    for (int j = 0; j < 4; ++j) {
        const int n  = bn + j * 16 + laneN;
        const float bv = bias[n];
#pragma unroll
        for (int r = 0; r < 8; ++r) {
            const int m  = bm + wid * 16 + r + laneH * 8;
            const float val = acc[j][r] + bv;
            if (SPLIT_HEADS) {
                const int b = m >> 12, s = m & (S_ - 1);
                const int h = n >> 6,  dk = n & (DK_ - 1);
                ((bf16_t*)Out)[(((size_t)(b * H_ + h) * S_ + s) << 6) + dk] = (bf16_t)val;
            } else {
                ((float*)Out)[(size_t)m * D_ + n] = val;
            }
        }
    }
}

__global__ __launch_bounds__(256)
void qkv_proj_kernel(const float* __restrict__ qx, const float* __restrict__ kx,
                     const float* __restrict__ vx,
                     const float* __restrict__ Wq, const float* __restrict__ bq,
                     const float* __restrict__ Wk, const float* __restrict__ bk,
                     const float* __restrict__ Wv, const float* __restrict__ bv,
                     bf16_t* __restrict__ qh, bf16_t* __restrict__ kh,
                     bf16_t* __restrict__ vh) {
    __shared__ __attribute__((aligned(16))) bf16_t As[128 * LDSA];
    __shared__ __attribute__((aligned(16))) bf16_t Bs[64 * LDSA];
    const int which = blockIdx.z;
    const float* X = (which == 0) ? qx : (which == 1) ? kx : vx;
    const float* W = (which == 0) ? Wq : (which == 1) ? Wk : Wv;
    const float* b = (which == 0) ? bq : (which == 1) ? bk : bv;
    bf16_t* out    = (which == 0) ? qh : (which == 1) ? kh : vh;
    gemm_body<false, true>(X, W, b, out, As, Bs);
}

__global__ __launch_bounds__(256)
void out_proj_kernel(const bf16_t* __restrict__ A, const float* __restrict__ Wo,
                     const float* __restrict__ bo, float* __restrict__ out) {
    __shared__ __attribute__((aligned(16))) bf16_t As[128 * LDSA];
    __shared__ __attribute__((aligned(16))) bf16_t Bs[64 * LDSA];
    gemm_body<true, false>(A, Wo, bo, out, As, Bs);
}

// ---------------------------------------------------------------------------
// Flash attention: grid (S/64, B*H), 128 thr = 4 waves; each wave owns 16 q rows.
// KV chunks of 32 staged by TDM into double-buffered LDS; online softmax in f32.
// ---------------------------------------------------------------------------
__global__ __launch_bounds__(128)
void flash_kernel(const bf16_t* __restrict__ qh, const bf16_t* __restrict__ kh,
                  const bf16_t* __restrict__ vh, bf16_t* __restrict__ concat) {
    __shared__ __attribute__((aligned(16))) bf16_t Ks[2][32 * LDSK];
    __shared__ __attribute__((aligned(16))) bf16_t Vs[2][32 * LDSK];
    __shared__ __attribute__((aligned(16))) bf16_t Ps[4 * 16 * LDSA];

    const int tid   = threadIdx.x;
    const int wid   = tid >> 5;
    const int lane  = tid & 31;
    const int laneN = lane & 15;
    const int laneH = lane >> 4;
    const int kb    = laneH * 8;
    const int bh    = blockIdx.y;                 // b*H + h
    const int qbase = blockIdx.x * 64 + wid * 16; // this wave's q rows
    const size_t base = (size_t)bh * S_ * DK_;

    // Q fragments for K-steps d=0..31 and d=32..63 (loaded once)
    const bf16_t* qrow = qh + base + (size_t)(qbase + laneN) * DK_;
    bf16x16 aq0 = make_frag(qrow + kb,      qrow + kb + 16);
    bf16x16 aq1 = make_frag(qrow + 32 + kb, qrow + 32 + kb + 16);

    float m_i[8], l_i[8];
    f32x8 o[4] = {};
#pragma unroll
    for (int r = 0; r < 8; ++r) { m_i[r] = -1e30f; l_i[r] = 0.0f; }

    // TDM prologue: chunk 0 -> buffer 0 (rows of 64 elems = 128B = 32 DW:
    // pad_interval code 4, pad 4 DW -> 144B row stride = LDSK elems)
    if (wid == 0) {
        tdm_load_2d((unsigned)(size_t)(void*)&Ks[0][0], kh + base,
                    64, 32, DK_, S_, DK_, 4, 3);
        tdm_load_2d((unsigned)(size_t)(void*)&Vs[0][0], vh + base,
                    64, 32, DK_, S_, DK_, 4, 3);
    }

    for (int j0 = 0; j0 < S_; j0 += 32) {
        const int buf = (j0 >> 5) & 1;
        if (wid == 0) {
            if (j0 + 32 < S_) {
                // prefetch next chunk into the other buffer, then wait so that
                // only the 2 just-issued ops remain outstanding (in-order cnt).
                tdm_load_2d((unsigned)(size_t)(void*)&Ks[buf ^ 1][0],
                            kh + base + (size_t)(j0 + 32) * DK_,
                            64, 32, DK_, S_, DK_, 4, 3);
                tdm_load_2d((unsigned)(size_t)(void*)&Vs[buf ^ 1][0],
                            vh + base + (size_t)(j0 + 32) * DK_,
                            64, 32, DK_, S_, DK_, 4, 3);
                __builtin_amdgcn_s_wait_tensorcnt(2);
            } else {
                __builtin_amdgcn_s_wait_tensorcnt(0);
            }
        }
        __syncthreads();   // current buffer ready for all waves

        // scores S[16 x 32] = Q(16x64) . K^T : two N-tiles, two K-steps each
        f32x8 s0 = {}, s1 = {};
        {
            const bf16_t* kr0 = &Ks[buf][(laneN) * LDSK];
            const bf16_t* kr1 = &Ks[buf][(16 + laneN) * LDSK];
            bf16x16 b00 = make_frag(kr0 + kb,      kr0 + kb + 16);
            bf16x16 b01 = make_frag(kr0 + 32 + kb, kr0 + 32 + kb + 16);
            bf16x16 b10 = make_frag(kr1 + kb,      kr1 + kb + 16);
            bf16x16 b11 = make_frag(kr1 + 32 + kb, kr1 + 32 + kb + 16);
            s0 = wmma_bf16(aq0, b00, s0);
            s0 = wmma_bf16(aq1, b01, s0);
            s1 = wmma_bf16(aq0, b10, s1);
            s1 = wmma_bf16(aq1, b11, s1);
        }

        // online softmax (rows live in 16-lane halves per C-layout)
        const float scale = 0.125f; // 1/sqrt(64)
        float p0[8], p1[8];
#pragma unroll
        for (int r = 0; r < 8; ++r) {
            float v0 = s0[r] * scale, v1 = s1[r] * scale;
            float mx = fmaxf(v0, v1);
#pragma unroll
            for (int off = 1; off < 16; off <<= 1)
                mx = fmaxf(mx, __shfl_xor(mx, off, 32));
            const float mnew  = fmaxf(m_i[r], mx);
            const float alpha = __expf(m_i[r] - mnew);
            const float e0 = __expf(v0 - mnew);
            const float e1 = __expf(v1 - mnew);
            float rs = e0 + e1;
#pragma unroll
            for (int off = 1; off < 16; off <<= 1)
                rs += __shfl_xor(rs, off, 32);
            l_i[r] = l_i[r] * alpha + rs;
            m_i[r] = mnew;
            p0[r] = e0; p1[r] = e1;
#pragma unroll
            for (int c = 0; c < 4; ++c) o[c][r] *= alpha;
        }

        // repack P (C layout) -> A-fragment layout via per-wave LDS region.
        // Wave-private region; LDS ops from one wave are in-order, no barrier.
        bf16_t* pw = &Ps[wid * 16 * LDSA];
#pragma unroll
        for (int r = 0; r < 8; ++r) {
            const int row = r + laneH * 8;
            pw[row * LDSA + laneN]      = (bf16_t)p0[r];
            pw[row * LDSA + 16 + laneN] = (bf16_t)p1[r];
        }

        bf16x16 ap = make_frag(&pw[laneN * LDSA + kb], &pw[laneN * LDSA + kb + 16]);
#pragma unroll
        for (int c = 0; c < 4; ++c) {
            const int n = c * 16 + laneN;
            bf16x16 bv;
#pragma unroll
            for (int i = 0; i < 8; ++i) {
                bv[i]     = Vs[buf][(kb + i) * LDSK + n];
                bv[8 + i] = Vs[buf][(kb + 16 + i) * LDSK + n];
            }
            o[c] = wmma_bf16(ap, bv, o[c]);
        }
        __syncthreads();   // release this buffer before TDM refills it (j0+64)
    }

    // normalize and store in concat layout [B, S, H, DK]
    const int b = bh >> 3, h = bh & (H_ - 1);
#pragma unroll
    for (int c = 0; c < 4; ++c) {
        const int dk = c * 16 + laneN;
#pragma unroll
        for (int r = 0; r < 8; ++r) {
            const int srow = qbase + r + laneH * 8;
            const float val = o[c][r] / l_i[r];
            concat[(((size_t)(b * S_ + srow) * H_ + h) << 6) + dk] = (bf16_t)val;
        }
    }
}

// ---------------------------------------------------------------------------
extern "C" void kernel_launch(void* const* d_in, const int* in_sizes, int n_in,
                              void* d_out, int out_size, void* d_ws, size_t ws_size,
                              hipStream_t stream) {
    (void)in_sizes; (void)n_in; (void)out_size; (void)ws_size;
    const float* qx = (const float*)d_in[0];
    const float* kx = (const float*)d_in[1];
    const float* vx = (const float*)d_in[2];
    const float* Wq = (const float*)d_in[3];
    const float* bq = (const float*)d_in[4];
    const float* Wk = (const float*)d_in[5];
    const float* bk = (const float*)d_in[6];
    const float* Wv = (const float*)d_in[7];
    const float* bv = (const float*)d_in[8];
    const float* Wo = (const float*)d_in[9];
    const float* bo = (const float*)d_in[10];

    const size_t per = (size_t)B_ * H_ * S_ * DK_; // 4,194,304 bf16 elements
    bf16_t* qh     = (bf16_t*)d_ws;
    bf16_t* kh     = qh + per;
    bf16_t* vh     = kh + per;
    bf16_t* concat = vh + per;

    qkv_proj_kernel<<<dim3(M_ / 128, D_ / 64, 3), 256, 0, stream>>>(
        qx, kx, vx, Wq, bq, Wk, bk, Wv, bv, qh, kh, vh);

    flash_kernel<<<dim3(S_ / 64, B_ * H_), 128, 0, stream>>>(qh, kh, vh, concat);

    out_proj_kernel<<<dim3(M_ / 128, D_ / 64), 256, 0, stream>>>(
        concat, Wo, bo, (float*)d_out);
}
